// MatchLSTM_17609365914096
// MI455X (gfx1250) — matrix operand order repaired and checked by reading, not compiled, not run on hardware
//
#include <hip/hip_runtime.h>
#include <math.h>

// ---------------------------------------------------------------------------
// MatchLSTM forward for MI455X (gfx1250, wave32, WMMA 16x16x32 f16 -> f32 acc)
// Round 2: register-blocked GEMMs (2x2 / 2x(3-gate)) for 2x less L2 traffic,
// plus global_prefetch on the streaming attention loop.
// ---------------------------------------------------------------------------

typedef __attribute__((ext_vector_type(16))) _Float16 v16h;
typedef __attribute__((ext_vector_type(8)))  _Float16 v8h;
typedef __attribute__((ext_vector_type(8)))  float    v8f;

union HFrag { v16h v; _Float16 h[16]; };
union FAcc  { v8f  v; float    f[8];  };

// A fragment: 16x32 (MxK) f16.  Lane L<16: row L, K = k0+0..7 and k0+16..23.
// Lane L>=16: row L-16, K = k0+8..15 and k0+24..31.  (CDNA5 ISA)
__device__ __forceinline__ v16h load_frag_a(const _Float16* __restrict__ base,
                                            int ld, int m0, int k0, int lane) {
  int row = m0 + (lane & 15);
  int kq  = (lane >> 4) ? 8 : 0;
  const _Float16* p = base + (size_t)row * ld + k0 + kq;
  HFrag f;
  *(v8h*)&f.h[0] = *(const v8h*)p;        // 16B vector load
  *(v8h*)&f.h[8] = *(const v8h*)(p + 16); // 16B vector load
  return f.v;
}

// B fragment: 32x16 (KxN) f16, W stored row-major [N,K] (i.e. B = W^T).
// Lane L<16: col L, K = k0+0..7 and k0+16..23 ; Lane L>=16: col L-16, K = k0+8..15 and k0+24..31.
__device__ __forceinline__ v16h load_frag_b(const _Float16* __restrict__ base,
                                            int ld, int n0, int k0, int lane) {
  int col = n0 + (lane & 15);
  int kb  = k0 + ((lane >> 4) ? 8 : 0);
  const _Float16* p = base + (size_t)col * ld + kb;
  HFrag f;
  *(v8h*)&f.h[0] = *(const v8h*)p;
  *(v8h*)&f.h[8] = *(const v8h*)(p + 16);
  return f.v;
}

__device__ __forceinline__ v8f wmma16(v16h a, v16h b, v8f c) {
  return __builtin_amdgcn_wmma_f32_16x16x32_f16(false, a, false, b, (short)0, c,
                                                false, false);
}

__device__ __forceinline__ float sigmoidf_(float x) {
  return 1.0f / (1.0f + expf(-x));
}

// ---------------------------------------------------------------------------
// Kernel: f32 -> f16 conversion with zero-padding of the K dimension
// ---------------------------------------------------------------------------
__global__ void cvt_f32_to_f16_pad(const float* __restrict__ src,
                                   _Float16* __restrict__ dst,
                                   int R, int Cs, int Cd) {
  int total = R * Cd;
  for (int i = blockIdx.x * blockDim.x + threadIdx.x; i < total;
       i += gridDim.x * blockDim.x) {
    int r = i / Cd, c = i % Cd;
    float v = (c < Cs) ? src[(size_t)r * Cs + c] : 0.0f;
    dst[i] = (_Float16)v;
  }
}

// ---------------------------------------------------------------------------
// Kernel: embedding gather -> padded f16 activation matrix [rows, Ep]
// ---------------------------------------------------------------------------
__global__ void embed_gather_f16(const int* __restrict__ idx,
                                 const float* __restrict__ embed,
                                 _Float16* __restrict__ out,
                                 int rows, int E, int Ep) {
  int total = rows * Ep;
  for (int i = blockIdx.x * blockDim.x + threadIdx.x; i < total;
       i += gridDim.x * blockDim.x) {
    int r = i / Ep, e = i % Ep;
    float v = (e < E) ? embed[(size_t)idx[r] * E + e] : 0.0f;
    out[i] = (_Float16)v;
  }
}

__global__ void zero_f16(_Float16* __restrict__ p, int n) {
  int i = blockIdx.x * blockDim.x + threadIdx.x;
  if (i < n) p[i] = (_Float16)0.0f;
}

// ---------------------------------------------------------------------------
// Kernel: GEMM  C[M,N] = A[M,K] * W[N,K]^T   (f16 in, f32 out)
// one wave per 32x32 tile (2x2 WMMA register blocking: each A fragment is
// reused across 2 B fragments and vice versa -> half the L2 traffic).
// grid = (M/32, N/32), block = 32
// ---------------------------------------------------------------------------
__global__ __launch_bounds__(32) void gemm_nt_f16(
    const _Float16* __restrict__ A, const _Float16* __restrict__ W,
    float* __restrict__ C, int M, int N, int K) {
  int m0 = blockIdx.x * 32, n0 = blockIdx.y * 32;
  int lane = threadIdx.x;
  v8f acc00 = {}, acc01 = {}, acc10 = {}, acc11 = {};
  for (int k0 = 0; k0 < K; k0 += 32) {
    v16h a0 = load_frag_a(A, K, m0,      k0, lane);
    v16h a1 = load_frag_a(A, K, m0 + 16, k0, lane);
    v16h b0 = load_frag_b(W, K, n0,      k0, lane);
    v16h b1 = load_frag_b(W, K, n0 + 16, k0, lane);
    acc00 = wmma16(a0, b0, acc00);
    acc01 = wmma16(a0, b1, acc01);
    acc10 = wmma16(a1, b0, acc10);
    acc11 = wmma16(a1, b1, acc11);
  }
  int col = lane & 15;
  int rb  = (lane >> 4) ? 8 : 0;
  FAcc r;
  r.v = acc00;
#pragma unroll
  for (int v = 0; v < 8; ++v) C[(size_t)(m0 + rb + v) * N + n0 + col] = r.f[v];
  r.v = acc01;
#pragma unroll
  for (int v = 0; v < 8; ++v) C[(size_t)(m0 + rb + v) * N + n0 + 16 + col] = r.f[v];
  r.v = acc10;
#pragma unroll
  for (int v = 0; v < 8; ++v) C[(size_t)(m0 + 16 + rb + v) * N + n0 + col] = r.f[v];
  r.v = acc11;
#pragma unroll
  for (int v = 0; v < 8; ++v) C[(size_t)(m0 + 16 + rb + v) * N + n0 + 16 + col] = r.f[v];
}

// ---------------------------------------------------------------------------
// Kernel: fused LSTMCell-from-zero-state GEMM.
//   g = X @ Wih^T + bih + bhh ; gates (i,f,g,o) at col offsets 0,H,2H,3H.
//   Zero cell state => forget gate is dead: compute only i,g,o.
//   Register blocking: 32 rows x 16 cols x 3 gates per wave -> 2 A fragments
//   reused across 3 gate B fragments (6 WMMAs per 5 fragment loads).
//   h = sigmoid(o)*tanh(sigmoid(i)*tanh(g)).
// grid = (M/32, H/16), block = 32
// ---------------------------------------------------------------------------
__global__ __launch_bounds__(32) void lstm_gate_gemm(
    const _Float16* __restrict__ X, const _Float16* __restrict__ Wih,
    const float* __restrict__ bih, const float* __restrict__ bhh,
    float* __restrict__ Hf32, _Float16* __restrict__ Hf16,
    int M, int K, int Hd) {
  int m0 = blockIdx.x * 32, n0 = blockIdx.y * 16;
  int lane = threadIdx.x;
  v8f ci0 = {}, cg0 = {}, co0 = {};
  v8f ci1 = {}, cg1 = {}, co1 = {};
  for (int k0 = 0; k0 < K; k0 += 32) {
    v16h a0 = load_frag_a(X, K, m0,      k0, lane);
    v16h a1 = load_frag_a(X, K, m0 + 16, k0, lane);
    v16h bi = load_frag_b(Wih, K, n0,          k0, lane);
    v16h bg = load_frag_b(Wih, K, n0 + 2 * Hd, k0, lane);
    v16h bo = load_frag_b(Wih, K, n0 + 3 * Hd, k0, lane);
    ci0 = wmma16(a0, bi, ci0);
    cg0 = wmma16(a0, bg, cg0);
    co0 = wmma16(a0, bo, co0);
    ci1 = wmma16(a1, bi, ci1);
    cg1 = wmma16(a1, bg, cg1);
    co1 = wmma16(a1, bo, co1);
  }
  int col = n0 + (lane & 15);
  int rb  = (lane >> 4) ? 8 : 0;
  float b_i = bih[col]          + bhh[col];
  float b_g = bih[col + 2 * Hd] + bhh[col + 2 * Hd];
  float b_o = bih[col + 3 * Hd] + bhh[col + 3 * Hd];
  FAcc ri, rg, ro;
#pragma unroll
  for (int half = 0; half < 2; ++half) {
    ri.v = half ? ci1 : ci0;
    rg.v = half ? cg1 : cg0;
    ro.v = half ? co1 : co0;
    int mbase = m0 + 16 * half + rb;
#pragma unroll
    for (int v = 0; v < 8; ++v) {
      float ig = sigmoidf_(ri.f[v] + b_i);
      float gg = tanhf(rg.f[v] + b_g);
      float og = sigmoidf_(ro.f[v] + b_o);
      float c  = ig * gg;
      float h  = og * tanhf(c);
      size_t o = (size_t)(mbase + v) * Hd + col;
      if (Hf32) Hf32[o] = h;
      Hf16[o] = (_Float16)h;
    }
  }
}

// ---------------------------------------------------------------------------
// Kernel: additive-attention scores.  One wave per (t,b):
//   scores[t,b] = sum_h w_e[h]*tanh(pre_s[t,b,h] + pre_t[k,b,h] + hmWm[b,h])
// pre_s is streamed 64 times per launch -> prefetch ahead (global_prefetch).
// ---------------------------------------------------------------------------
__global__ __launch_bounds__(32) void attn_score(
    const float* __restrict__ pre_s, const float* __restrict__ pre_t_k,
    const float* __restrict__ hmWm, const float* __restrict__ w_e,
    float* __restrict__ scores, int B, int Hd) {
  int tb = blockIdx.x;           // t*B + b
  int b  = tb % B;
  int lane = threadIdx.x;
  const float* ps = pre_s   + (size_t)tb * Hd;
  const float* pt = pre_t_k + (size_t)b  * Hd;
  const float* hw = hmWm    + (size_t)b  * Hd;
  float acc = 0.0f;
  for (int h = lane; h < Hd; h += 32) {
    __builtin_prefetch(ps + h + 128, 0, 1);   // -> global_prefetch_b8
    acc += w_e[h] * tanhf(ps[h] + pt[h] + hw[h]);
  }
#pragma unroll
  for (int o = 16; o; o >>= 1) acc += __shfl_xor(acc, o, 32);
  if (lane == 0) scores[tb] = acc;
}

// ---------------------------------------------------------------------------
// Kernel: softmax over the premise axis (T=64) for each batch column b.
// grid = B, block = T
// ---------------------------------------------------------------------------
__global__ __launch_bounds__(64) void softmax_col(
    const float* __restrict__ scores, float* __restrict__ alpha, int T, int B) {
  int b = blockIdx.x, t = threadIdx.x;
  __shared__ float sh[64];
  float v = scores[t * B + b];
  sh[t] = v; __syncthreads();
  for (int s = 32; s; s >>= 1) { if (t < s) sh[t] = fmaxf(sh[t], sh[t + s]); __syncthreads(); }
  float m = sh[0]; __syncthreads();
  float e = expf(v - m);
  sh[t] = e; __syncthreads();
  for (int s = 32; s; s >>= 1) { if (t < s) sh[t] += sh[t + s]; __syncthreads(); }
  alpha[t * B + b] = e / sh[0];
}

// ---------------------------------------------------------------------------
// Kernel: a_k[b,h] = sum_t alpha[t,b]*h_s[t,b,h]; build m_k = [a_k | h_t_k]
// (f16, K-major for the following WMMA GEMM).  grid = B, block = 256
// ---------------------------------------------------------------------------
__global__ __launch_bounds__(256) void ak_concat(
    const float* __restrict__ alpha, const _Float16* __restrict__ hs16,
    const _Float16* __restrict__ ht16_k, _Float16* __restrict__ mk,
    int T, int B, int Hd) {
  int b = blockIdx.x;
  for (int h = threadIdx.x; h < Hd; h += blockDim.x) {
    float acc = 0.0f;
    for (int t = 0; t < T; ++t)
      acc += alpha[t * B + b] * (float)hs16[((size_t)t * B + b) * Hd + h];
    mk[(size_t)b * 2 * Hd + h]      = (_Float16)acc;
    mk[(size_t)b * 2 * Hd + Hd + h] = ht16_k[(size_t)b * Hd + h];
  }
}

// ---------------------------------------------------------------------------
// Kernel: final classifier  out[b,c] = h_m[b,:] . fc_w[c,:] + fc_b[c]
// ---------------------------------------------------------------------------
__global__ void fc_out(const float* __restrict__ h, const float* __restrict__ w,
                       const float* __restrict__ bias, float* __restrict__ out,
                       int B, int C, int Hd) {
  int i = blockIdx.x * blockDim.x + threadIdx.x;
  if (i >= B * C) return;
  int b = i / C, c = i % C;
  float acc = bias[c];
  for (int k = 0; k < Hd; ++k)
    acc += h[(size_t)b * Hd + k] * w[(size_t)c * Hd + k];
  out[i] = acc;
}

// ---------------------------------------------------------------------------
// Host driver
// ---------------------------------------------------------------------------
static inline size_t align256(size_t x) { return (x + 255) & ~(size_t)255; }

extern "C" void kernel_launch(void* const* d_in, const int* in_sizes, int n_in,
                              void* d_out, int out_size, void* d_ws, size_t ws_size,
                              hipStream_t stream) {
  (void)in_sizes; (void)n_in; (void)out_size; (void)ws_size;
  const int TP = 64, TH = 64, Bn = 128, Hd = 512, E = 300, Ep = 320, Cc = 3;
  const int MP = TP * Bn;               // 8192 premise rows
  const int MH = TH * Bn;               // 8192 hypothesis rows
  const int H4 = 4 * Hd;                // 2048
  const int K2 = 2 * Hd;                // 1024

  const int*   premise    = (const int*)  d_in[0];
  const int*   hypothesis = (const int*)  d_in[2];
  const float* embed      = (const float*)d_in[4];
  const float* w_e        = (const float*)d_in[5];
  const float* Ws         = (const float*)d_in[6];
  const float* Wt         = (const float*)d_in[7];
  const float* Wm         = (const float*)d_in[8];
  const float* Wih_p      = (const float*)d_in[9];
  const float* bih_p      = (const float*)d_in[10];
  const float* bhh_p      = (const float*)d_in[11];
  const float* Wih_h      = (const float*)d_in[12];
  const float* bih_h      = (const float*)d_in[13];
  const float* bhh_h      = (const float*)d_in[14];
  const float* Wih_m      = (const float*)d_in[15];
  const float* bih_m      = (const float*)d_in[16];
  const float* bhh_m      = (const float*)d_in[17];
  const float* fc_w       = (const float*)d_in[18];
  const float* fc_b       = (const float*)d_in[19];
  float* out = (float*)d_out;

  // ---- carve workspace ----
  char* ws = (char*)d_ws;
  size_t off = 0;
  auto carve = [&](size_t bytes) -> void* {
    void* p = ws + off; off = align256(off + bytes); return p;
  };
  _Float16* Xp16    = (_Float16*)carve((size_t)MP * Ep * 2);
  _Float16* Xh16    = (_Float16*)carve((size_t)MH * Ep * 2);
  _Float16* Wihp16  = (_Float16*)carve((size_t)H4 * Ep * 2);
  _Float16* Wihh16  = (_Float16*)carve((size_t)H4 * Ep * 2);
  _Float16* Wihm16  = (_Float16*)carve((size_t)H4 * K2 * 2);
  _Float16* Ws16    = (_Float16*)carve((size_t)Hd * Hd * 2);
  _Float16* Wt16    = (_Float16*)carve((size_t)Hd * Hd * 2);
  _Float16* Wm16    = (_Float16*)carve((size_t)Hd * Hd * 2);
  _Float16* hs16    = (_Float16*)carve((size_t)MP * Hd * 2);
  _Float16* ht16    = (_Float16*)carve((size_t)MH * Hd * 2);
  float*    pre_s   = (float*)   carve((size_t)MP * Hd * 4);
  float*    pre_t   = (float*)   carve((size_t)MH * Hd * 4);
  float*    hmWm    = (float*)   carve((size_t)Bn * Hd * 4);
  float*    scores  = (float*)   carve((size_t)TP * Bn * 4);
  float*    alpha   = (float*)   carve((size_t)TP * Bn * 4);
  _Float16* mk16    = (_Float16*)carve((size_t)Bn * K2 * 2);
  float*    hm      = (float*)   carve((size_t)Bn * Hd * 4);
  _Float16* hm16    = (_Float16*)carve((size_t)Bn * Hd * 2);

  // ---- one-time weight conversion (f32 -> f16, pad E 300->320) ----
  auto cvt = [&](const float* s, _Float16* d, int R, int Cs, int Cd) {
    int n = R * Cd;
    cvt_f32_to_f16_pad<<<(n + 255) / 256, 256, 0, stream>>>(s, d, R, Cs, Cd);
  };
  cvt(Wih_p, Wihp16, H4, E, Ep);
  cvt(Wih_h, Wihh16, H4, E, Ep);
  cvt(Wih_m, Wihm16, H4, K2, K2);
  cvt(Ws, Ws16, Hd, Hd, Hd);
  cvt(Wt, Wt16, Hd, Hd, Hd);
  cvt(Wm, Wm16, Hd, Hd, Hd);

  // ---- embedding gathers ----
  {
    int n = MP * Ep;
    embed_gather_f16<<<(n + 255) / 256, 256, 0, stream>>>(premise, embed, Xp16, MP, E, Ep);
    n = MH * Ep;
    embed_gather_f16<<<(n + 255) / 256, 256, 0, stream>>>(hypothesis, embed, Xh16, MH, E, Ep);
  }

  // ---- encoder LSTM cells (zero state; fused gate GEMM + activation) ----
  {
    dim3 grid(MP / 32, Hd / 16);
    lstm_gate_gemm<<<grid, 32, 0, stream>>>(Xp16, Wihp16, bih_p, bhh_p,
                                            (float*)nullptr, hs16, MP, Ep, Hd);
    lstm_gate_gemm<<<grid, 32, 0, stream>>>(Xh16, Wihh16, bih_h, bhh_h,
                                            (float*)nullptr, ht16, MH, Ep, Hd);
  }

  // ---- pre_s = h_s @ Ws^T, pre_t = h_t @ Wt^T ----
  {
    dim3 grid(MP / 32, Hd / 32);
    gemm_nt_f16<<<grid, 32, 0, stream>>>(hs16, Ws16, pre_s, MP, Hd, Hd);
    gemm_nt_f16<<<grid, 32, 0, stream>>>(ht16, Wt16, pre_t, MH, Hd, Hd);
  }

  // ---- init h_m = 0 ----
  {
    int n = Bn * Hd;
    zero_f16<<<(n + 255) / 256, 256, 0, stream>>>(hm16, n);
  }

  // ---- match-LSTM scan over hypothesis timesteps ----
  dim3 gridHmGemm(Bn / 32, Hd / 32);   // h_m @ Wm^T
  dim3 gridHmGate(Bn / 32, Hd / 16);   // match cell gates
  for (int k = 0; k < TH; ++k) {
    // h_m_prev @ Wm^T
    gemm_nt_f16<<<gridHmGemm, 32, 0, stream>>>(hm16, Wm16, hmWm, Bn, Hd, Hd);
    // attention scores (one wave per (t,b))
    attn_score<<<TP * Bn, 32, 0, stream>>>(pre_s, pre_t + (size_t)k * Bn * Hd,
                                           hmWm, w_e, scores, Bn, Hd);
    // softmax over premise axis
    softmax_col<<<Bn, TP, 0, stream>>>(scores, alpha, TP, Bn);
    // a_k and concat -> m_k
    ak_concat<<<Bn, 256, 0, stream>>>(alpha, hs16, ht16 + (size_t)k * Bn * Hd,
                                      mk16, TP, Bn, Hd);
    // match cell (zero state) -> h_m
    lstm_gate_gemm<<<gridHmGate, 32, 0, stream>>>(mk16, Wihm16, bih_m, bhh_m,
                                                  hm, hm16, Bn, K2, Hd);
  }

  // ---- classifier ----
  fc_out<<<(Bn * Cc + 255) / 256, 256, 0, stream>>>(hm, fc_w, fc_b, out, Bn, Cc, Hd);
}
